// GENConvNN_57354993270880
// MI455X (gfx1250) — matrix operand order repaired and checked by reading, not compiled, not run on hardware
//
#include <hip/hip_runtime.h>

typedef __bf16 bf16_t;
typedef __bf16 v16bf __attribute__((ext_vector_type(16)));
typedef __bf16 v8bf  __attribute__((ext_vector_type(8)));
typedef float  v8f   __attribute__((ext_vector_type(8)));

#define HID 128
#define H2  256

// ---------------- zero workspace ----------------
__global__ void zero_kernel(float* __restrict__ p, long n) {
    long i = (long)blockIdx.x * blockDim.x + threadIdx.x;
    long stride = (long)gridDim.x * blockDim.x;
    for (; i < n; i += stride) p[i] = 0.0f;
}

// ---------------- weight transpose+cast: src[K][Nout] f32 -> dst[Nout][K] bf16 ----------------
__global__ void wtprep_kernel(const float* __restrict__ src, bf16_t* __restrict__ dst,
                              int K, int Nout) {
    long total = (long)K * Nout;
    long i = (long)blockIdx.x * blockDim.x + threadIdx.x;
    long stride = (long)gridDim.x * blockDim.x;
    for (; i < total; i += stride) {
        int k = (int)(i / Nout);
        int n = (int)(i % Nout);
        dst[(long)n * K + k] = (bf16_t)src[i];
    }
}

// ---------------- embed: h = relu(LN(x @ w_emb + b)) ; 16 nodes / 128-thread block ----------------
__global__ void embed_kernel(const float* __restrict__ x, const float* __restrict__ w_emb,
                             const float* __restrict__ b_emb, const float* __restrict__ ln_g,
                             const float* __restrict__ ln_b, float* __restrict__ h, int N) {
    __shared__ float xs[16][36];
    __shared__ float hb[16][HID];
    __shared__ float mu_s[16], rs_s[16];
    const int t = threadIdx.x;            // 128 threads
    const int nb = blockIdx.x * 16;
    for (int i = t; i < 16 * 36; i += 128) xs[i / 36][i % 36] = x[(long)nb * 36 + i];
    __syncthreads();
    float acc[16];
#pragma unroll
    for (int i = 0; i < 16; ++i) acc[i] = 0.0f;
    for (int k = 0; k < 36; ++k) {
        float wv = w_emb[k * HID + t];
#pragma unroll
        for (int i = 0; i < 16; ++i) acc[i] += xs[i][k] * wv;
    }
    float be = b_emb[t];
#pragma unroll
    for (int i = 0; i < 16; ++i) hb[i][t] = acc[i] + be;
    __syncthreads();
    if (t < 16) {
        float s = 0.0f, q = 0.0f;
        for (int k = 0; k < HID; ++k) { float v = hb[t][k]; s += v; q += v * v; }
        float mu = s * (1.0f / HID);
        float var = q * (1.0f / HID) - mu * mu;
        mu_s[t] = mu;
        rs_s[t] = rsqrtf(var + 1e-5f);
    }
    __syncthreads();
    float g = ln_g[t], bb = ln_b[t];
#pragma unroll
    for (int i = 0; i < 16; ++i) {
        float v = (hb[i][t] - mu_s[i]) * rs_s[i] * g + bb;
        h[(long)(nb + i) * HID + t] = fmaxf(v, 0.0f);
    }
}

// ---------------- edge pass 1: m = relu(h[src]+lin_edge(e))+eps ; atomic max into mmax[dst] ----------------
__global__ void edge_max_kernel(const float* __restrict__ edge_attr, const float* __restrict__ w_edge,
                                const float* __restrict__ b_edge, const float* __restrict__ h,
                                const int* __restrict__ ei, int* __restrict__ mmax_i, int E) {
    long idx = (long)blockIdx.x * blockDim.x + threadIdx.x;
    if (idx >= (long)E * HID) return;
    int e = (int)(idx >> 7);
    int f = (int)(idx & 127);
    int src = ei[e];
    int dst = ei[E + e];
    float ea0 = edge_attr[e * 4 + 0], ea1 = edge_attr[e * 4 + 1];
    float ea2 = edge_attr[e * 4 + 2], ea3 = edge_attr[e * 4 + 3];
    float ev = ea0 * w_edge[0 * HID + f] + ea1 * w_edge[1 * HID + f] +
               ea2 * w_edge[2 * HID + f] + ea3 * w_edge[3 * HID + f] + b_edge[f];
    float m = fmaxf(h[(long)src * HID + f] + ev, 0.0f) + 1e-7f;   // m > 0 always
    atomicMax(mmax_i + (long)dst * HID + f, __float_as_int(m));   // positive floats: int order == float order
}

// ---------------- edge pass 2: ex = exp(m - mmax[dst]); denom += ex; num += m*ex ----------------
__global__ void edge_sum_kernel(const float* __restrict__ edge_attr, const float* __restrict__ w_edge,
                                const float* __restrict__ b_edge, const float* __restrict__ h,
                                const int* __restrict__ ei, const float* __restrict__ mmax,
                                float* __restrict__ denom, float* __restrict__ num, int E) {
    long idx = (long)blockIdx.x * blockDim.x + threadIdx.x;
    if (idx >= (long)E * HID) return;
    int e = (int)(idx >> 7);
    int f = (int)(idx & 127);
    int src = ei[e];
    int dst = ei[E + e];
    float ea0 = edge_attr[e * 4 + 0], ea1 = edge_attr[e * 4 + 1];
    float ea2 = edge_attr[e * 4 + 2], ea3 = edge_attr[e * 4 + 3];
    float ev = ea0 * w_edge[0 * HID + f] + ea1 * w_edge[1 * HID + f] +
               ea2 * w_edge[2 * HID + f] + ea3 * w_edge[3 * HID + f] + b_edge[f];
    float m = fmaxf(h[(long)src * HID + f] + ev, 0.0f) + 1e-7f;
    float ex = __expf(m - mmax[(long)dst * HID + f]);
    atomicAdd(denom + (long)dst * HID + f, ex);
    atomicAdd(num + (long)dst * HID + f, m * ex);
}

// ---------------- combine: a = h + num/(denom+1e-16) -> bf16 [N,128] ----------------
__global__ void combine_kernel(const float* __restrict__ h, const float* __restrict__ num,
                               const float* __restrict__ denom, bf16_t* __restrict__ a_bf, long total) {
    long i = (long)blockIdx.x * blockDim.x + threadIdx.x;
    if (i >= total) return;
    float aggr = num[i] / (denom[i] + 1e-16f);
    a_bf[i] = (bf16_t)(h[i] + aggr);
}

// ---------------- bf16 WMMA GEMM: Z[N,Nout] = A[N,K] @ Wt^T + bias ; optional stats / relu ----------------
// block = 256 threads = 8 waves, tiles 64 rows x 32 cols; each wave one 16x16 tile via v_wmma_f32_16x16x32_bf16
__global__ void gemm_kernel(const bf16_t* __restrict__ A, const bf16_t* __restrict__ Wt,
                            const float* __restrict__ bias, bf16_t* __restrict__ Z,
                            float* __restrict__ gsum, float* __restrict__ gsumsq,
                            int N, int K, int Nout, int do_relu) {
    const int tid  = threadIdx.x;
    const int w    = tid >> 5;          // wave 0..7
    const int lane = tid & 31;
    const int half = lane >> 4;         // 0: lanes 0-15, 1: lanes 16-31
    const int l16  = lane & 15;
    const int wm = w >> 1, wn = w & 1;
    const int rowbase = blockIdx.x * 64 + wm * 16;
    const int colbase = blockIdx.y * 32 + wn * 16;

    int arow = rowbase + l16;
    int arow_c = arow < N ? arow : (N - 1);           // OOB rows only affect OOB C rows (unstored)
    const bf16_t* aptr = A + (long)arow_c * K;
    const bf16_t* bptr = Wt + (long)(colbase + l16) * K + half * 16;

    v8f c = {};
    for (int k0 = 0; k0 < K; k0 += 32) {
        // A 16x32 bf16 fragment: lane half h holds 16 consecutive K at k0+16h
        v8bf a0 = *(const v8bf*)(aptr + k0 + half * 16);
        v8bf a1 = *(const v8bf*)(aptr + k0 + half * 16 + 8);
        // B 32x16 bf16 fragment: lane half h holds 16 consecutive K at k0+16h
        v8bf b0 = *(const v8bf*)(bptr + k0);
        v8bf b1 = *(const v8bf*)(bptr + k0 + 8);
        v16bf af, bfv;
#pragma unroll
        for (int i = 0; i < 8; ++i) {
            af[i] = a0[i]; af[i + 8] = a1[i];
            bfv[i] = b0[i]; bfv[i + 8] = b1[i];
        }
        c = __builtin_amdgcn_wmma_f32_16x16x32_bf16(false, af, false, bfv,
                                                    (short)0, c, false, false);
    }

    const float bcol = bias[colbase + l16];
    float s = 0.0f, q = 0.0f;
#pragma unroll
    for (int v = 0; v < 8; ++v) {
        int row = rowbase + half * 8 + v;             // C layout: lane half h -> M = v + 8h
        float val = c[v] + bcol;
        if (do_relu) val = fmaxf(val, 0.0f);
        if (row < N) {
            Z[(long)row * Nout + colbase + l16] = (bf16_t)val;
            s += val;
            q += val * val;
        }
    }
    if (gsum != nullptr) {
        s += __shfl_xor(s, 16);                       // combine rows 0-7 with rows 8-15 (same column)
        q += __shfl_xor(q, 16);
        if (half == 0) {
            atomicAdd(gsum + colbase + l16, s);
            atomicAdd(gsumsq + colbase + l16, q);
        }
    }
}

// ---------------- fold BN stats into per-feature scale/shift ----------------
__global__ void bn_prep_kernel(const float* __restrict__ gsum, const float* __restrict__ gsumsq,
                               const float* __restrict__ g, const float* __restrict__ b,
                               float* __restrict__ scale, float* __restrict__ shift, int Nn) {
    int f = threadIdx.x;
    if (f >= H2) return;
    float inv = 1.0f / (float)Nn;
    float mu = gsum[f] * inv;
    float var = gsumsq[f] * inv - mu * mu;
    float sc = g[f] * rsqrtf(var + 1e-5f);
    scale[f] = sc;
    shift[f] = b[f] - mu * sc;
}

// ---------------- apply BN affine + relu: a = relu(z*scale + shift) (bf16 -> bf16) ----------------
__global__ void bn_relu_kernel(const bf16_t* __restrict__ z, const float* __restrict__ scale,
                               const float* __restrict__ shift, bf16_t* __restrict__ a, long total) {
    long i = (long)blockIdx.x * blockDim.x + threadIdx.x;
    if (i >= total) return;
    int f = (int)(i & (H2 - 1));
    float v = (float)z[i] * scale[f] + shift[f];
    a[i] = (bf16_t)fmaxf(v, 0.0f);
}

// ---------------- decoder: out[N,12] = act[N,128] @ w_dec + b_dec ; 64 nodes / 256-thread block ----------------
__global__ void dec_kernel(const bf16_t* __restrict__ act, const float* __restrict__ w_dec,
                           const float* __restrict__ b_dec, float* __restrict__ out, int N) {
    __shared__ float  wd[HID * 12];
    __shared__ bf16_t as[64 * HID];
    const int t = threadIdx.x;           // 256
    for (int i = t; i < HID * 12; i += 256) wd[i] = w_dec[i];
    const int nb = blockIdx.x * 64;
    const int rows = (N - nb) < 64 ? (N - nb) : 64;
    const unsigned int* asrc = (const unsigned int*)(act + (long)nb * HID);
    unsigned int* adst = (unsigned int*)as;
    for (int i = t; i < rows * (HID / 2); i += 256) adst[i] = asrc[i];
    __syncthreads();
    const int node = t >> 2, grp = t & 3;            // each thread: 3 outputs of one node
    if (node < rows) {
        float a0 = b_dec[grp * 3 + 0], a1 = b_dec[grp * 3 + 1], a2 = b_dec[grp * 3 + 2];
        for (int k = 0; k < HID; ++k) {
            float av = (float)as[node * HID + k];
            a0 += av * wd[k * 12 + grp * 3 + 0];
            a1 += av * wd[k * 12 + grp * 3 + 1];
            a2 += av * wd[k * 12 + grp * 3 + 2];
        }
        long o = (long)(nb + node) * 12 + grp * 3;
        out[o + 0] = a0; out[o + 1] = a1; out[o + 2] = a2;
    }
}

extern "C" void kernel_launch(void* const* d_in, const int* in_sizes, int n_in,
                              void* d_out, int out_size, void* d_ws, size_t ws_size,
                              hipStream_t stream) {
    const float* x         = (const float*)d_in[0];
    const float* edge_attr = (const float*)d_in[1];
    const float* w_emb     = (const float*)d_in[2];
    const float* b_emb     = (const float*)d_in[3];
    const float* ln_g      = (const float*)d_in[4];
    const float* ln_b      = (const float*)d_in[5];
    const float* w_edge    = (const float*)d_in[6];
    const float* b_edge    = (const float*)d_in[7];
    const float* wmlp[5]   = { (const float*)d_in[8],  (const float*)d_in[10],
                               (const float*)d_in[12], (const float*)d_in[14],
                               (const float*)d_in[16] };
    const float* bmlp[5]   = { (const float*)d_in[9],  (const float*)d_in[11],
                               (const float*)d_in[13], (const float*)d_in[15],
                               (const float*)d_in[17] };
    const float* bn_g      = (const float*)d_in[18];
    const float* bn_b      = (const float*)d_in[19];
    const float* w_dec     = (const float*)d_in[20];
    const float* b_dec     = (const float*)d_in[21];
    const int*   ei        = (const int*)d_in[22];

    const int N = in_sizes[0] / 36;
    const int E = in_sizes[1] / 4;

    // ---- workspace layout (f32 message buffers aliased to bf16 activation ping-pong) ----
    char* ws = (char*)d_ws;
    const size_t szN128 = (size_t)N * HID * sizeof(float);     // == N*256*sizeof(bf16)
    float*  h     = (float*)(ws);
    float*  mmax  = (float*)(ws + 1 * szN128);
    float*  denom = (float*)(ws + 2 * szN128);
    float*  num   = (float*)(ws + 3 * szN128);
    bf16_t* a_bf  = (bf16_t*)mmax;    // alias: free after edge pass 2
    bf16_t* z_bf  = (bf16_t*)denom;   // alias: free after combine
    bf16_t* wt1   = (bf16_t*)(ws + 4 * szN128);
    bf16_t* wt2   = wt1 + 256 * 128;
    bf16_t* wt3   = wt2 + 256 * 256;
    bf16_t* wt4   = wt3 + 256 * 256;
    bf16_t* wt5   = wt4 + 256 * 256;
    float*  stats = (float*)(wt5 + 128 * 256);                 // 4 layers * {sum,sumsq,scale,shift}*256

    // ---- zero message buffers + stats (required each call: atomics accumulate) ----
    {
        long n = 3L * N * HID;
        zero_kernel<<<4096, 256, 0, stream>>>(mmax, n);
        zero_kernel<<<16, 256, 0, stream>>>(stats, 4L * 4 * 256);
    }

    // ---- transpose+cast MLP weights to [Nout][K] bf16 ----
    wtprep_kernel<<<128, 256, 0, stream>>>(wmlp[0], wt1, 128, 256);
    wtprep_kernel<<<128, 256, 0, stream>>>(wmlp[1], wt2, 256, 256);
    wtprep_kernel<<<128, 256, 0, stream>>>(wmlp[2], wt3, 256, 256);
    wtprep_kernel<<<128, 256, 0, stream>>>(wmlp[3], wt4, 256, 256);
    wtprep_kernel<<<128, 256, 0, stream>>>(wmlp[4], wt5, 256, 128);

    // ---- embed ----
    embed_kernel<<<(N + 15) / 16, 128, 0, stream>>>(x, w_emb, b_emb, ln_g, ln_b, h, N);

    // ---- edge softmax aggregation (2 passes, recompute m; hot set ~L2 resident) ----
    {
        long total = (long)E * HID;
        int blocks = (int)((total + 255) / 256);
        edge_max_kernel<<<blocks, 256, 0, stream>>>(edge_attr, w_edge, b_edge, h, ei, (int*)mmax, E);
        edge_sum_kernel<<<blocks, 256, 0, stream>>>(edge_attr, w_edge, b_edge, h, ei, mmax, denom, num, E);
    }

    // ---- combine: a_bf = bf16(h + aggr)  [N,128] ----
    {
        long total = (long)N * HID;
        combine_kernel<<<(int)((total + 255) / 256), 256, 0, stream>>>(h, num, denom, a_bf, total);
    }

    const int gx = (N + 63) / 64;
    long totH2 = (long)N * H2;
    int bnBlocks = (int)((totH2 + 255) / 256);

    // ---- MLP layers 1..4: GEMM(+stats) -> bn_prep -> bn_relu ----
    for (int L = 0; L < 4; ++L) {
        const bf16_t* Wt = (L == 0) ? wt1 : (L == 1) ? wt2 : (L == 2) ? wt3 : wt4;
        int K = (L == 0) ? 128 : 256;
        float* gs  = stats + L * 1024;
        float* gq  = gs + 256;
        float* gsc = gs + 512;
        float* gsh = gs + 768;
        gemm_kernel<<<dim3(gx, 8), 256, 0, stream>>>(a_bf, Wt, bmlp[L], z_bf, gs, gq, N, K, 256, 0);
        bn_prep_kernel<<<1, 256, 0, stream>>>(gs, gq, bn_g + L * H2, bn_b + L * H2, gsc, gsh, N);
        bn_relu_kernel<<<bnBlocks, 256, 0, stream>>>(z_bf, gsc, gsh, a_bf, totH2);
    }

    // ---- layer 5: [N,256]@[256,128] + relu (no BN) -> z region [N,128] bf16 ----
    gemm_kernel<<<dim3(gx, 4), 256, 0, stream>>>(a_bf, wt5, bmlp[4], z_bf, nullptr, nullptr, N, 256, 128, 1);

    // ---- decoder -> d_out [N,6,2] f32 ----
    dec_kernel<<<(N + 63) / 64, 256, 0, stream>>>(z_bf, w_dec, b_dec, (float*)d_out, N);
}
